// FP8MoEModuleForInputScaleTest_20839181320306
// MI455X (gfx1250) — 8-bit matrix operand order repaired and checked by reading, compiled, not run on hardware
//
#include <hip/hip_runtime.h>
#include <hip/hip_bf16.h>
#include <stdint.h>

static constexpr int T = 4096;
static constexpr int H = 2048;
static constexpr int I = 4096;
static constexpr int E = 8;
static constexpr int TOPK = 2;

typedef __attribute__((ext_vector_type(16))) int   v16i;
typedef __attribute__((ext_vector_type(8)))  float v8f;
typedef int v4i_b __attribute__((vector_size(16)));

static constexpr size_t OFF_COUNTS = 0;
static constexpr size_t OFF_TOK    = 256;
static constexpr size_t OFF_GW     = OFF_TOK + (size_t)E * T * 4;
static constexpr size_t OFF_W1Q    = OFF_GW  + (size_t)E * T * 4;
static constexpr size_t OFF_W2Q    = OFF_W1Q + (size_t)E * 2 * I * H;
static constexpr size_t OFF_XQ     = OFF_W2Q + (size_t)E * H * I;
static constexpr size_t OFF_AQ     = OFF_XQ  + (size_t)E * T * H;

__device__ __forceinline__ float clamp_fp8(float x) {
    return fminf(448.0f, fmaxf(-448.0f, x));
}

__device__ __forceinline__ unsigned pack4_fp8_raw(float a, float b, float c, float d) {
    int r = __builtin_amdgcn_cvt_pk_fp8_f32(a, b, 0, false);
    r     = __builtin_amdgcn_cvt_pk_fp8_f32(c, d, r, true);
    return (unsigned)r;
}

__device__ __forceinline__ unsigned pack4_fp8_clamped(float a, float b, float c, float d) {
    return pack4_fp8_raw(clamp_fp8(a), clamp_fp8(b), clamp_fp8(c), clamp_fp8(d));
}

__device__ __forceinline__ unsigned char cvt1_fp8(float a) {
    float c = clamp_fp8(a);
    return (unsigned char)(__builtin_amdgcn_cvt_pk_fp8_f32(c, c, 0, false) & 0xFF);
}

__device__ __forceinline__ void async_cp16(const unsigned char* g, unsigned char* l) {
    __builtin_amdgcn_global_load_async_to_lds_b128(
        (__attribute__((address_space(1))) v4i_b*)g,
        (__attribute__((address_space(3))) v4i_b*)l, 0, 0);
}

__device__ __forceinline__ v8f wmma_fp8(v16i a, v16i b, v8f c) {
    return __builtin_amdgcn_wmma_f32_16x16x128_fp8_fp8(a, b, (short)0, c, false, false);
}

__device__ __forceinline__ v16i load_a_frag(const unsigned char* As, int mBase, int lane) {
    const int pat[8] = {0, 4, 8, 12, 32, 36, 40, 44};
    const unsigned char* p = As + (mBase + (lane & 15)) * 128 + ((lane >> 4) << 4);
    v16i a;
#pragma unroll
    for (int i = 0; i < 8; ++i) {
        a[i]     = *(const int*)(p + pat[i]);
        a[i + 8] = *(const int*)(p + pat[i] + 64);
    }
    return a;
}

__device__ __forceinline__ v16i load_b_frag(const unsigned char* Bs, int nBase, int lane) {
    const unsigned char* p = Bs + (nBase + (lane & 15)) * 128 + ((lane >> 4) << 4);
    v16i b;
#pragma unroll
    for (int g = 0; g < 4; ++g) {
        uint4 q = *(const uint4*)(p + g * 32);
        b[4 * g + 0] = (int)q.x; b[4 * g + 1] = (int)q.y;
        b[4 * g + 2] = (int)q.z; b[4 * g + 3] = (int)q.w;
    }
    return b;
}

__global__ __launch_bounds__(256) void zero_out_kernel(float* out, size_t n, int* counts) {
    size_t i      = (size_t)blockIdx.x * blockDim.x + threadIdx.x;
    size_t stride = (size_t)gridDim.x * blockDim.x;
    for (size_t k = i; k < n; k += stride) out[k] = 0.0f;
    if (blockIdx.x == 0 && threadIdx.x < E) counts[threadIdx.x] = 0;
}

__global__ __launch_bounds__(256) void route_kernel(const int* __restrict__ sel,
                                                    const float* __restrict__ rw,
                                                    int* counts, int* tok, float* gw) {
    int i = blockIdx.x * blockDim.x + threadIdx.x;
    if (i >= T * TOPK) return;
    int   t = i / TOPK;
    int   e = sel[i];
    float w = rw[i];
    int pos = atomicAdd(&counts[e], 1);
    if (pos < T) {
        tok[e * T + pos] = t;
        gw[e * T + pos]  = w;
    }
}

__global__ __launch_bounds__(256) void quant_w_kernel(const float* __restrict__ w,
                                                      unsigned char* __restrict__ wq,
                                                      size_t n4) {
    size_t i      = (size_t)blockIdx.x * blockDim.x + threadIdx.x;
    size_t stride = (size_t)gridDim.x * blockDim.x;
    const float4* s = (const float4*)w;
    unsigned*     d = (unsigned*)wq;
    for (; i < n4; i += stride) {
        float4 v = s[i];
        d[i] = pack4_fp8_raw(v.x, v.y, v.z, v.w);
    }
}

__global__ __launch_bounds__(256) void gather_quant_x_kernel(const float* __restrict__ x,
                                                             const int* __restrict__ counts,
                                                             const int* __restrict__ tok,
                                                             const float* __restrict__ w1_iscale,
                                                             unsigned char* __restrict__ xq) {
    int e    = blockIdx.y;
    int slot = blockIdx.x;
    int cnt  = counts[e]; if (cnt > T) cnt = T;
    if (slot >= cnt) return;
    int   t   = tok[e * T + slot];
    float inv = 1.0f / w1_iscale[e];
    const float4* src = (const float4*)(x + (size_t)t * H);
    unsigned*     dst = (unsigned*)(xq + ((size_t)e * T + slot) * H);
    for (int i = threadIdx.x; i < H / 4; i += 256) {
        float4 v = src[i];
        dst[i] = pack4_fp8_clamped(v.x * inv, v.y * inv, v.z * inv, v.w * inv);
    }
}

__global__ __launch_bounds__(256) void moe_gemm1_kernel(const unsigned char* __restrict__ xq,
                                                        const unsigned char* __restrict__ w1q,
                                                        const int* __restrict__ counts,
                                                        unsigned char* __restrict__ aq,
                                                        const float* __restrict__ w1_iscale,
                                                        const float* __restrict__ w1_wscale,
                                                        const float* __restrict__ w2_iscale) {
    const int e  = blockIdx.z;
    int cnt = counts[e]; if (cnt > T) cnt = T;
    const int m0 = blockIdx.y * 128;
    if (m0 >= cnt) return;
    const int j0 = blockIdx.x * 64;

    __shared__ unsigned char As[2][128 * 128];
    __shared__ unsigned char Bs[2][128 * 128];

    const int tid  = threadIdx.x;
    const int lane = tid & 31;
    const int wave = tid >> 5;
    const int mw   = wave & 3;
    const int nw   = wave >> 2;

    const int ldRow = tid >> 1;
    const int ldC0  = (tid & 1) * 64;
    int ar = m0 + ldRow; if (ar >= cnt) ar = cnt - 1;
    const unsigned char* asrc = xq + ((size_t)e * T + ar) * H + ldC0;
    const size_t wrow = (ldRow < 64) ? (size_t)(j0 + ldRow)
                                     : (size_t)I + (size_t)(j0 + ldRow - 64);
    const unsigned char* bsrc = w1q + ((size_t)e * (2 * I) + wrow) * H + ldC0;

    v8f accg[2][2] = {};
    v8f accu[2][2] = {};

    constexpr int NIT = H / 128;

    {
        unsigned char* al = &As[0][ldRow * 128 + ldC0];
        unsigned char* bl = &Bs[0][ldRow * 128 + ldC0];
#pragma unroll
        for (int i = 0; i < 4; ++i) async_cp16(asrc + i * 16, al + i * 16);
#pragma unroll
        for (int i = 0; i < 4; ++i) async_cp16(bsrc + i * 16, bl + i * 16);
    }

    for (int it = 0; it < NIT; ++it) {
        const int cur = it & 1;
        if (it + 1 < NIT) {
            const int nxt = cur ^ 1;
            const int kk  = (it + 1) * 128;
            unsigned char* al = &As[nxt][ldRow * 128 + ldC0];
            unsigned char* bl = &Bs[nxt][ldRow * 128 + ldC0];
#pragma unroll
            for (int i = 0; i < 4; ++i) async_cp16(asrc + kk + i * 16, al + i * 16);
#pragma unroll
            for (int i = 0; i < 4; ++i) async_cp16(bsrc + kk + i * 16, bl + i * 16);
            __builtin_amdgcn_s_wait_asynccnt(8);
        } else {
            __builtin_amdgcn_s_wait_asynccnt(0);
        }
        __syncthreads();

        v16i a0 = load_a_frag(As[cur], mw * 32,      lane);
        v16i a1 = load_a_frag(As[cur], mw * 32 + 16, lane);
#pragma unroll
        for (int ct = 0; ct < 2; ++ct) {
            v16i bg = load_b_frag(Bs[cur],      nw * 32 + ct * 16, lane);
            v16i bu = load_b_frag(Bs[cur], 64 + nw * 32 + ct * 16, lane);
            accg[0][ct] = wmma_fp8(a0, bg, accg[0][ct]);
            accg[1][ct] = wmma_fp8(a1, bg, accg[1][ct]);
            accu[0][ct] = wmma_fp8(a0, bu, accu[0][ct]);
            accu[1][ct] = wmma_fp8(a1, bu, accu[1][ct]);
        }
        __syncthreads();
    }

    const float s1      = w1_iscale[e] * w1_wscale[e];
    const float inv_is2 = 1.0f / w2_iscale[e];
    const int   voff    = (lane >> 4) * 8;
    const int   ncol    = lane & 15;
#pragma unroll
    for (int mt = 0; mt < 2; ++mt) {
#pragma unroll
        for (int ct = 0; ct < 2; ++ct) {
            const int col = j0 + nw * 32 + ct * 16 + ncol;
#pragma unroll
            for (int v = 0; v < 8; ++v) {
                const int r = m0 + mw * 32 + mt * 16 + voff + v;
                if (r < cnt) {
                    float g = accg[mt][ct][v] * s1;
                    float u = accu[mt][ct][v] * s1;
                    float a = (g / (1.0f + __expf(-g))) * u;
                    aq[((size_t)e * T + r) * I + col] = cvt1_fp8(a * inv_is2);
                }
            }
        }
    }
}

__global__ __launch_bounds__(256) void moe_gemm2_kernel(const unsigned char* __restrict__ aq,
                                                        const unsigned char* __restrict__ w2q,
                                                        const int* __restrict__ counts,
                                                        const int* __restrict__ tok,
                                                        const float* __restrict__ gw,
                                                        float* __restrict__ out,
                                                        const float* __restrict__ w2_iscale,
                                                        const float* __restrict__ w2_wscale) {
    const int e  = blockIdx.z;
    int cnt = counts[e]; if (cnt > T) cnt = T;
    const int m0 = blockIdx.y * 128;
    if (m0 >= cnt) return;
    const int h0 = blockIdx.x * 128;

    __shared__ unsigned char As[2][128 * 128];
    __shared__ unsigned char Bs[2][128 * 128];
    __shared__ int           stok[128];
    __shared__ float         sgw[128];

    const int tid  = threadIdx.x;
    const int lane = tid & 31;
    const int wave = tid >> 5;
    const int mw   = wave & 3;
    const int nw   = wave >> 2;

    if (tid < 128) {
        int r = m0 + tid;
        stok[tid] = (r < cnt) ? tok[e * T + r] : 0;
        sgw[tid]  = (r < cnt) ? gw[e * T + r] : 0.0f;
    }

    const int ldRow = tid >> 1;
    const int ldC0  = (tid & 1) * 64;
    int ar = m0 + ldRow; if (ar >= cnt) ar = cnt - 1;
    const unsigned char* asrc = aq + ((size_t)e * T + ar) * I + ldC0;
    const unsigned char* bsrc = w2q + ((size_t)e * H + (h0 + ldRow)) * I + ldC0;

    v8f acc[2][4] = {};

    constexpr int NIT = I / 128;

    {
        unsigned char* al = &As[0][ldRow * 128 + ldC0];
        unsigned char* bl = &Bs[0][ldRow * 128 + ldC0];
#pragma unroll
        for (int i = 0; i < 4; ++i) async_cp16(asrc + i * 16, al + i * 16);
#pragma unroll
        for (int i = 0; i < 4; ++i) async_cp16(bsrc + i * 16, bl + i * 16);
    }

    for (int it = 0; it < NIT; ++it) {
        const int cur = it & 1;
        if (it + 1 < NIT) {
            const int nxt = cur ^ 1;
            const int kk  = (it + 1) * 128;
            unsigned char* al = &As[nxt][ldRow * 128 + ldC0];
            unsigned char* bl = &Bs[nxt][ldRow * 128 + ldC0];
#pragma unroll
            for (int i = 0; i < 4; ++i) async_cp16(asrc + kk + i * 16, al + i * 16);
#pragma unroll
            for (int i = 0; i < 4; ++i) async_cp16(bsrc + kk + i * 16, bl + i * 16);
            __builtin_amdgcn_s_wait_asynccnt(8);
        } else {
            __builtin_amdgcn_s_wait_asynccnt(0);
        }
        __syncthreads();

        v16i a0 = load_a_frag(As[cur], mw * 32,      lane);
        v16i a1 = load_a_frag(As[cur], mw * 32 + 16, lane);
#pragma unroll
        for (int ct = 0; ct < 4; ++ct) {
            v16i b = load_b_frag(Bs[cur], nw * 64 + ct * 16, lane);
            acc[0][ct] = wmma_fp8(a0, b, acc[0][ct]);
            acc[1][ct] = wmma_fp8(a1, b, acc[1][ct]);
        }
        __syncthreads();
    }

    const float s2   = w2_iscale[e] * w2_wscale[e];
    const int   voff = (lane >> 4) * 8;
    const int   ncol = lane & 15;
#pragma unroll
    for (int mt = 0; mt < 2; ++mt) {
#pragma unroll
        for (int ct = 0; ct < 4; ++ct) {
            const int col = h0 + nw * 64 + ct * 16 + ncol;
#pragma unroll
            for (int v = 0; v < 8; ++v) {
                const int rl = mw * 32 + mt * 16 + voff + v;
                if (m0 + rl < cnt) {
                    float y = acc[mt][ct][v] * s2;
                    atomicAdd(out + (size_t)stok[rl] * H + col, sgw[rl] * y);
                }
            }
        }
    }
}

extern "C" void kernel_launch(void* const* d_in, const int* in_sizes, int n_in,
                              void* d_out, int out_size, void* d_ws, size_t ws_size,
                              hipStream_t stream) {
    const float* x     = (const float*)d_in[0];
    const int*   sel   = (const int*)d_in[1];
    const float* rw    = (const float*)d_in[2];
    const float* w1    = (const float*)d_in[3];
    const float* w2    = (const float*)d_in[4];
    const float* w1_is = (const float*)d_in[5];
    const float* w2_is = (const float*)d_in[6];
    const float* w1_ws = (const float*)d_in[7];
    const float* w2_ws = (const float*)d_in[8];
    float*       out   = (float*)d_out;

    char* ws = (char*)d_ws;
    int*           counts = (int*)(ws + OFF_COUNTS);
    int*           tok    = (int*)(ws + OFF_TOK);
    float*         gw     = (float*)(ws + OFF_GW);
    unsigned char* w1q    = (unsigned char*)(ws + OFF_W1Q);
    unsigned char* w2q    = (unsigned char*)(ws + OFF_W2Q);
    unsigned char* xq     = (unsigned char*)(ws + OFF_XQ);
    unsigned char* aq     = (unsigned char*)(ws + OFF_AQ);

    zero_out_kernel<<<2048, 256, 0, stream>>>(out, (size_t)T * H, counts);
    route_kernel<<<(T * TOPK + 255) / 256, 256, 0, stream>>>(sel, rw, counts, tok, gw);
    quant_w_kernel<<<4096, 256, 0, stream>>>(w1, w1q, (size_t)E * 2 * I * H / 4);
    quant_w_kernel<<<2048, 256, 0, stream>>>(w2, w2q, (size_t)E * H * I / 4);
    gather_quant_x_kernel<<<dim3(T, E), 256, 0, stream>>>(x, counts, tok, w1_is, xq);
    moe_gemm1_kernel<<<dim3(I / 64, T / 128, E), 256, 0, stream>>>(xq, w1q, counts, aq,
                                                                   w1_is, w1_ws, w2_is);
    moe_gemm2_kernel<<<dim3(H / 128, T / 128, E), 256, 0, stream>>>(aq, w2q, counts, tok, gw,
                                                                    out, w2_is, w2_ws);
}
